// FastConformerAttention_18004502905001
// MI455X (gfx1250) — matrix operand order repaired and checked by reading, not compiled, not run on hardware
//
#include <hip/hip_runtime.h>
#include <hip/hip_bf16.h>

typedef __bf16 bf16;
typedef __bf16 v16bf __attribute__((ext_vector_type(16)));
typedef float  v8f   __attribute__((ext_vector_type(8)));
typedef unsigned int v4u __attribute__((ext_vector_type(4)));
typedef int v8i __attribute__((ext_vector_type(8)));
typedef int v4i __attribute__((ext_vector_type(4)));

#define WMMA_BF16(a, b, c) \
    __builtin_amdgcn_wmma_f32_16x16x32_bf16(false, (a), false, (b), (short)0, (c), false, false)

union FragCvt { v16bf v; uint4 q[2]; };

// A-matrix bf16 16x32 fragment (CDNA5 layout):
// lane 0-15 holds row r, K in {k0..k0+7, k0+16..k0+23};
// lane 16-31 holds row r, K in {k0+8..k0+15, k0+24..k0+31}.
static __device__ __forceinline__ v16bf load_a(const bf16* __restrict__ base, int ld,
                                               int row0, int k0, int rmax) {
    int lane = threadIdx.x & 31;
    int r = lane & 15, hf = lane >> 4;
    int row = row0 + r;
    row = row > rmax ? rmax : row;
    const bf16* p = base + (size_t)row * ld + k0 + hf * 8;
    FragCvt u;
    u.q[0] = *(const uint4*)(p);
    u.q[1] = *(const uint4*)(p + 16);
    return u.v;
}

// B-matrix bf16 32x16 fragment: lane n (0-15) holds column n with K in {k0..k0+7, k0+16..k0+23},
// lane 16-31 holds column n with K in {k0+8..k0+15, k0+24..k0+31}.
// "column n" is row n of the row-major (N,K) source matrix.
static __device__ __forceinline__ v16bf load_b(const bf16* __restrict__ base, int ld,
                                               int col0, int k0, int cmax) {
    int lane = threadIdx.x & 31;
    int n = lane & 15, hf = lane >> 4;
    int row = col0 + n;
    row = row > cmax ? cmax : row;
    row = row < 0 ? 0 : row;
    const bf16* p = base + (size_t)row * ld + k0 + hf * 8;
    FragCvt u;
    u.q[0] = *(const uint4*)(p);
    u.q[1] = *(const uint4*)(p + 16);
    return u.v;
}

// ---- Tensor Data Mover: 2D bf16 tile (tile_w x tile_h, row pitch in elems) -> LDS.
// D# packed per CDNA5 ISA; 6-arg builtin (clang-23 flavor).
static __device__ __forceinline__ void tdm_load_tile_bf16(const bf16* gsrc, unsigned lds_off,
                                                          int tile_w, int tile_h,
                                                          int pitch_elems) {
    unsigned long long ga = (unsigned long long)(size_t)gsrc;
    v4u g0;
    g0[0] = 1u;                                   // count=1, is_restore=0, gather=0
    g0[1] = lds_off;                              // lds_addr (bytes)
    g0[2] = (unsigned)ga;                         // global_addr[31:0]
    g0[3] = (unsigned)((ga >> 32) & 0x01FFFFFFu) | (2u << 30);  // addr[56:32] | type=2
    v8i g1;
    g1[0] = (int)(1u << 16);                      // workgroup_mask=0, data_size=1 (2B)
    g1[1] = (int)((unsigned)(tile_w & 0xFFFF) << 16);           // tensor_dim0[15:0]
    g1[2] = (int)(((unsigned)tile_w >> 16) & 0xFFFFu) |
            (int)((unsigned)(tile_h & 0xFFFF) << 16);           // td0 hi | tensor_dim1 lo
    g1[3] = (int)(((unsigned)tile_h >> 16) & 0xFFFFu) |
            (int)((unsigned)(tile_w & 0xFFFF) << 16);           // td1 hi | tile_dim0
    g1[4] = (int)(tile_h & 0xFFFF);               // tile_dim1 | tile_dim2=0
    g1[5] = pitch_elems;                          // tensor_dim0_stride[31:0]
    g1[6] = 0;                                    // stride hi | dim1_stride lo
    g1[7] = 0;
    v4i z4 = {0, 0, 0, 0};
    v8i z8 = {0, 0, 0, 0, 0, 0, 0, 0};
    __builtin_amdgcn_tensor_load_to_lds(g0, g1, z4, z4, z8, 0);
}

// ---------------------------------------------------------------- convert
__global__ void cvt_f32_bf16(const float* __restrict__ src, bf16* __restrict__ dst, long n) {
    long i = (long)blockIdx.x * blockDim.x + threadIdx.x;
    long stride = (long)gridDim.x * blockDim.x;
    for (; i < n; i += stride) dst[i] = (bf16)src[i];
}

// ---------------------------------------------------------------- GEMM: C = A @ Bw^T (+colbias)
// A: (M,K) bf16 row-major.  Bw: (N,K) bf16 row-major.
// mode 0: bf16 row-major out (ld=N).  mode 1: fp32 row-major out.
// mode 2: bf16 out transposed per head -> [B][H=8][DK=128][Tdim] (for V).
__global__ __launch_bounds__(256) void gemm_bf16_kernel(
    const bf16* __restrict__ A, const bf16* __restrict__ Bw, void* __restrict__ Cout,
    const float* __restrict__ colbias, int M, int N, int K, int mode, int Tdim) {
    int w = threadIdx.x >> 5, lane = threadIdx.x & 31;
    int wm = w >> 1, wn = w & 1;
    int m0 = blockIdx.y * 128 + wm * 32;
    int n0 = blockIdx.x * 128 + wn * 64;
    v8f acc[2][4] = {};

    for (int kb = 0; kb < K; kb += 32) {
        v16bf af[2];
#pragma unroll
        for (int i = 0; i < 2; ++i) af[i] = load_a(A, K, m0 + i * 16, kb, M - 1);
#pragma unroll
        for (int j = 0; j < 4; ++j) {
            v16bf bfr = load_b(Bw, K, n0 + j * 16, kb, N - 1);
#pragma unroll
            for (int i = 0; i < 2; ++i) acc[i][j] = WMMA_BF16(af[i], bfr, acc[i][j]);
        }
    }

    int r16 = lane & 15, hf = lane >> 4;
#pragma unroll
    for (int i = 0; i < 2; ++i)
#pragma unroll
        for (int j = 0; j < 4; ++j) {
            int col = n0 + j * 16 + r16;
            float bias = colbias ? colbias[col] : 0.f;
#pragma unroll
            for (int r = 0; r < 8; ++r) {
                int row = m0 + i * 16 + r + 8 * hf;
                if (row < M) {
                    float val = acc[i][j][r] + bias;
                    if (mode == 0) {
                        ((bf16*)Cout)[(size_t)row * N + col] = (bf16)val;
                    } else if (mode == 1) {
                        ((float*)Cout)[(size_t)row * N + col] = val;
                    } else {  // V transposed per head
                        int b = row / Tdim, t = row % Tdim;
                        int hh = col >> 7, d = col & 127;
                        ((bf16*)Cout)[(size_t)((b * 8 + hh) * 128 + d) * Tdim + t] = (bf16)val;
                    }
                }
            }
        }
}

// ---------------------------------------------------------------- fused rel-pos flash attention
// Qu/Qv/Kb: (B*T, D) bf16.  Vt: [B][H][DK][T] bf16.  Pp: (2047 pad 2048, D) bf16.
// O: (B*T, D) bf16.  One block = (b, h, 128 q rows); 8 waves, 16 q rows each.
// K and V^T tiles double-buffered in LDS via the Tensor Data Mover (DMA for tile i+1
// overlaps compute on tile i); P band streamed from L2.
__global__ __launch_bounds__(256) void attn_relpos_kernel(
    const bf16* __restrict__ Qu, const bf16* __restrict__ Qv, const bf16* __restrict__ Kb,
    const bf16* __restrict__ Vt, const bf16* __restrict__ Pp, bf16* __restrict__ O) {
    const int T = 1024, D = 1024, DK = 128, PMAX = 2046;
    int q0 = blockIdx.x * 128;
    int h = blockIdx.y;
    int b = blockIdx.z;
    int w = threadIdx.x >> 5, lane = threadIdx.x & 31;
    int r16 = lane & 15, hf = lane >> 4;
    int qb = q0 + w * 16;

    const size_t rowbase = (size_t)b * T;
    const bf16* QuH = Qu + rowbase * D + h * DK;
    const bf16* QvH = Qv + rowbase * D + h * DK;
    const bf16* KH  = Kb + rowbase * D + h * DK;
    const bf16* PH  = Pp + h * DK;
    const bf16* VH  = Vt + (size_t)(b * 8 + h) * DK * T;

    __shared__ __align__(16) bf16 kbuf[2][128 * 128];  // K tiles: rows j-local, cols d
    __shared__ __align__(16) bf16 vbuf[2][128 * 128];  // V^T tiles: rows d, cols j-local
    // per-wave scratch: rel-shift (16x48 f32 = 3KB) and exp(S) (16x128 bf16 = 4KB)
    // are temporally disjoint within a j-tile -> overlay in 4KB.
    __shared__ __align__(16) unsigned char wscratch[8][4096];

    // Q fragments are invariant over the whole j loop: keep in registers.
    v16bf au[4], av[4];
#pragma unroll
    for (int c = 0; c < 4; ++c) {
        au[c] = load_a(QuH, D, qb, c * 32, T - 1);
        av[c] = load_a(QvH, D, qb, c * 32, T - 1);
    }

    v8f o[8] = {};
    float mi[8], li[8];
#pragma unroll
    for (int r = 0; r < 8; ++r) { mi[r] = -1e30f; li[r] = 0.f; }

    // Prime the pipeline: DMA tile 0 into buffer 0.
    if (w == 0) {
        tdm_load_tile_bf16(KH, (unsigned)(size_t)&kbuf[0][0], 128, 128, D);
        tdm_load_tile_bf16(VH, (unsigned)(size_t)&vbuf[0][0], 128, 128, T);
    }

    for (int j0 = 0; j0 < T; j0 += 128) {
        int ib = (j0 >> 7) & 1;
        // --- issue DMA for tile i+1 into the other buffer, then wait for tile i only.
        if (w == 0) {
            bool more = (j0 + 128) < T;
            if (more) {
                tdm_load_tile_bf16(KH + (size_t)(j0 + 128) * D,
                                   (unsigned)(size_t)&kbuf[ib ^ 1][0], 128, 128, D);
                tdm_load_tile_bf16(VH + (j0 + 128),
                                   (unsigned)(size_t)&vbuf[ib ^ 1][0], 128, 128, T);
                __builtin_amdgcn_s_wait_tensorcnt((short)2);  // tile i done, i+1 in flight
            } else {
                __builtin_amdgcn_s_wait_tensorcnt((short)0);
            }
        }
        __syncthreads();

        const bf16* kb_ = kbuf[ib];
        const bf16* vb_ = vbuf[ib];
        float* sw = (float*)wscratch[w];
        v8f s[8];
#pragma unroll
        for (int jn = 0; jn < 8; ++jn) {
            // --- AC: (q+u) . k   (K fragments from LDS)
            v8f acc = {};
#pragma unroll
            for (int c = 0; c < 4; ++c) {
                v16bf bk = load_b(kb_, 128, jn * 16, c * 32, 127);
                acc = WMMA_BF16(au[c], bk, acc);
            }
            // --- BD: (q+v) . p[T-1-q+j] via 16x32 band GEMM + skewed LDS shift
            v8f sf0 = {}, sf1 = {};
            int prow = (T - 1) - qb + j0 + jn * 16 - 15;
#pragma unroll
            for (int c = 0; c < 4; ++c) {
                v16bf p0 = load_b(PH, D, prow, c * 32, PMAX);
                v16bf p1 = load_b(PH, D, prow + 16, c * 32, PMAX);
                sf0 = WMMA_BF16(av[c], p0, sf0);
                sf1 = WMMA_BF16(av[c], p1, sf1);
            }
#pragma unroll
            for (int r = 0; r < 8; ++r) {
                int qr = r + 8 * hf;
                sw[qr * 48 + r16 + qr]      = sf0[r];  // skew: col = t + qr
                sw[qr * 48 + r16 + 16 + qr] = sf1[r];
            }
            asm volatile("s_wait_dscnt 0x0" ::: "memory");
#pragma unroll
            for (int r = 0; r < 8; ++r) {
                int qr = r + 8 * hf;
                s[jn][r] = acc[r] + sw[qr * 48 + 15 + r16];  // uniform read = shifted BD
            }
        }
        // --- online softmax (rows live on 16-lane halves; width-16 shuffles)
        const float scale = 0.08838834764831845f;  // 1/sqrt(128)
        float alpha[8];
#pragma unroll
        for (int r = 0; r < 8; ++r) {
            float mt = -1e30f;
#pragma unroll
            for (int jn = 0; jn < 8; ++jn) { s[jn][r] *= scale; mt = fmaxf(mt, s[jn][r]); }
#pragma unroll
            for (int off = 1; off < 16; off <<= 1) mt = fmaxf(mt, __shfl_xor(mt, off, 16));
            float mnew = fmaxf(mi[r], mt);
            alpha[r] = __expf(mi[r] - mnew);
            mi[r] = mnew;
            float rs = 0.f;
#pragma unroll
            for (int jn = 0; jn < 8; ++jn) {
                float e = __expf(s[jn][r] - mnew);
                s[jn][r] = e;
                rs += e;
            }
#pragma unroll
            for (int off = 1; off < 16; off <<= 1) rs += __shfl_xor(rs, off, 16);
            li[r] = li[r] * alpha[r] + rs;
        }
#pragma unroll
        for (int nd = 0; nd < 8; ++nd)
#pragma unroll
            for (int r = 0; r < 8; ++r) o[nd][r] *= alpha[r];
        // --- stage exp(S) as bf16 A-matrix in (overlaid) LDS scratch
        asm volatile("s_wait_dscnt 0x0" ::: "memory");
        bf16* pw = (bf16*)wscratch[w];
#pragma unroll
        for (int jn = 0; jn < 8; ++jn)
#pragma unroll
            for (int r = 0; r < 8; ++r) {
                int qr = r + 8 * hf;
                pw[qr * 128 + jn * 16 + r16] = (bf16)s[jn][r];
            }
        asm volatile("s_wait_dscnt 0x0" ::: "memory");
        // --- O += P @ V   (both operands from LDS)
#pragma unroll
        for (int nd = 0; nd < 8; ++nd)
#pragma unroll
            for (int c = 0; c < 4; ++c) {
                v16bf a = load_a(pw, 128, 0, c * 32, 15);
                v16bf bv = load_b(vb_, 128, nd * 16, c * 32, 127);
                o[nd] = WMMA_BF16(a, bv, o[nd]);
            }
        __syncthreads();  // buffer fully consumed before TDM overwrites it next iter
    }
    // epilogue: normalize and store (B,T,D) with out[b,q,h,d] layout
    bf16* Orow = O + rowbase * D + h * DK;
#pragma unroll
    for (int r = 0; r < 8; ++r) {
        int qr = r + 8 * hf;
        float inv = 1.f / li[r];
#pragma unroll
        for (int nd = 0; nd < 8; ++nd)
            Orow[(size_t)(qb + qr) * D + nd * 16 + r16] = (bf16)(o[nd][r] * inv);
    }
}

// ---------------------------------------------------------------- host launcher
extern "C" void kernel_launch(void* const* d_in, const int* in_sizes, int n_in,
                              void* d_out, int out_size, void* d_ws, size_t ws_size,
                              hipStream_t stream) {
    const float* X   = (const float*)d_in[0];  // (4,1024,1024)
    const float* pos = (const float*)d_in[1];  // (1,2047,1024)
    const float* Wq  = (const float*)d_in[2];
    const float* Wk  = (const float*)d_in[3];
    const float* Wv  = (const float*)d_in[4];
    const float* Wo  = (const float*)d_in[5];
    const float* Wp  = (const float*)d_in[6];
    const float* bu  = (const float*)d_in[7];  // (8,128) flat = per-column bias
    const float* bv  = (const float*)d_in[8];

    const long MT = 4096L * 1024;   // B*T x D
    const long WW = 1024L * 1024;
    const long PT = 2047L * 1024;
    const long PPAD = 2048L * 1024;

    bf16* ws = (bf16*)d_ws;
    size_t off = 0;
    bf16* Xb   = ws + off; off += MT;
    bf16* Wqb  = ws + off; off += WW;
    bf16* Wkb  = ws + off; off += WW;
    bf16* Wvb  = ws + off; off += WW;
    bf16* Wob  = ws + off; off += WW;
    bf16* Wpb  = ws + off; off += WW;
    bf16* Posb = ws + off; off += PPAD;
    bf16* Qu   = ws + off; off += MT;
    bf16* Qv   = ws + off; off += MT;
    bf16* Kb   = ws + off; off += MT;
    bf16* Vt   = ws + off; off += MT;   // [B][H][DK][T]
    bf16* Pp   = ws + off; off += PPAD;
    bf16* Ob   = ws + off; off += MT;

    // fp32 -> bf16 conversions
    cvt_f32_bf16<<<2048, 256, 0, stream>>>(X, Xb, MT);
    cvt_f32_bf16<<<2048, 256, 0, stream>>>(pos, Posb, PT);
    cvt_f32_bf16<<<512, 256, 0, stream>>>(Wq, Wqb, WW);
    cvt_f32_bf16<<<512, 256, 0, stream>>>(Wk, Wkb, WW);
    cvt_f32_bf16<<<512, 256, 0, stream>>>(Wv, Wvb, WW);
    cvt_f32_bf16<<<512, 256, 0, stream>>>(Wo, Wob, WW);
    cvt_f32_bf16<<<512, 256, 0, stream>>>(Wp, Wpb, WW);

    dim3 g4096(8, 32), gpos(8, 16), blk(256);
    // Qu = X Wq^T + u ; Qv = X Wq^T + v ; K = X Wk^T ; Vt = (X Wv^T)^T-per-head ; Pp = pos Wp^T
    gemm_bf16_kernel<<<g4096, blk, 0, stream>>>(Xb, Wqb, Qu, bu, 4096, 1024, 1024, 0, 1024);
    gemm_bf16_kernel<<<g4096, blk, 0, stream>>>(Xb, Wqb, Qv, bv, 4096, 1024, 1024, 0, 1024);
    gemm_bf16_kernel<<<g4096, blk, 0, stream>>>(Xb, Wkb, Kb, nullptr, 4096, 1024, 1024, 0, 1024);
    gemm_bf16_kernel<<<g4096, blk, 0, stream>>>(Xb, Wvb, Vt, nullptr, 4096, 1024, 1024, 2, 1024);
    gemm_bf16_kernel<<<gpos, blk, 0, stream>>>(Posb, Wpb, Pp, nullptr, 2047, 1024, 1024, 0, 1024);

    // fused relative-position flash attention (double-buffered TDM K/V tiles)
    attn_relpos_kernel<<<dim3(8, 8, 4), blk, 0, stream>>>(Qu, Qv, Kb, Vt, Pp, Ob);

    // final projection: out = O @ Wo^T (fp32 out)
    gemm_bf16_kernel<<<g4096, blk, 0, stream>>>(Ob, Wob, d_out, nullptr, 4096, 1024, 1024, 1, 1024);

    (void)in_sizes; (void)n_in; (void)out_size; (void)ws_size;
}
